// MambaformerBlock_40278203302043
// MI455X (gfx1250) — matrix operand order repaired and checked by reading, not compiled, not run on hardware
//
#include <hip/hip_runtime.h>
#include <hip/hip_bf16.h>
#include <math.h>

#define NTOK   8192   // B*L
#define DIMC   128
#define DINNER 256
#define DSTATE 128
#define LSEQ   2048
#define BATCH  4

typedef __attribute__((ext_vector_type(8)))  __bf16 v8bf;
typedef __attribute__((ext_vector_type(16))) __bf16 v16bf;
typedef __attribute__((ext_vector_type(8)))  float  v8f;

__device__ __forceinline__ float silu_f(float v) { return v / (1.0f + __expf(-v)); }

__device__ __forceinline__ v16bf cat16(v8bf a, v8bf b)
{
    return __builtin_shufflevector(a, b, 0,1,2,3,4,5,6,7,8,9,10,11,12,13,14,15);
}

// ---------------- f32 -> bf16 copy (weights, one-shot) ----------------
__global__ __launch_bounds__(256) void cvt_bf16_kernel(const float* __restrict__ s,
                                                       __bf16* __restrict__ d, int n)
{
    int i = blockIdx.x * blockDim.x + threadIdx.x;
    if (i < n) d[i] = (__bf16)s[i];
}

// ---------------- LayerNorm: one wave per 128-wide token row ----------------
template<bool BF16OUT>
__global__ __launch_bounds__(256) void ln_kernel(const float* __restrict__ x,
                                                 const float* __restrict__ g,
                                                 const float* __restrict__ bb,
                                                 float* __restrict__ out,
                                                 __bf16* __restrict__ outb, int ntok)
{
    int wave = (blockIdx.x * blockDim.x + threadIdx.x) >> 5;
    int lane = threadIdx.x & 31;
    if (wave >= ntok) return;
    const float* row = x + (size_t)wave * DIMC;
    float v0 = row[lane], v1 = row[lane + 32], v2 = row[lane + 64], v3 = row[lane + 96];
    float s = v0 + v1 + v2 + v3;
#pragma unroll
    for (int off = 16; off; off >>= 1) s += __shfl_xor(s, off, 32);
    float mu = s * (1.0f / 128.0f);
    float d0 = v0 - mu, d1 = v1 - mu, d2 = v2 - mu, d3 = v3 - mu;
    float vs = d0 * d0 + d1 * d1 + d2 * d2 + d3 * d3;
#pragma unroll
    for (int off = 16; off; off >>= 1) vs += __shfl_xor(vs, off, 32);
    float inv = rsqrtf(vs * (1.0f / 128.0f) + 1e-5f);
    float o0 = d0 * inv * g[lane]      + bb[lane];
    float o1 = d1 * inv * g[lane + 32] + bb[lane + 32];
    float o2 = d2 * inv * g[lane + 64] + bb[lane + 64];
    float o3 = d3 * inv * g[lane + 96] + bb[lane + 96];
    if (BF16OUT) {
        __bf16* orow = outb + (size_t)wave * DIMC;
        orow[lane] = (__bf16)o0; orow[lane + 32] = (__bf16)o1;
        orow[lane + 64] = (__bf16)o2; orow[lane + 96] = (__bf16)o3;
    } else {
        float* orow = out + (size_t)wave * DIMC;
        orow[lane] = o0; orow[lane + 32] = o1;
        orow[lane + 64] = o2; orow[lane + 96] = o3;
    }
}

// ---- GEMM: Y[m,n] = alpha*act(X[m,:]·W[n,:] + bias[n]) + res[m,n] ----------
// Pure-bf16 fragments (preconverted in memory), f32 accumulate.
// Each wave: 32(M) x 64(N) strip -> 8 accumulators; per K-step: 2 A-frags +
// 4 B-frags (12x b128 loads) feed 8 WMMAs with 2x fragment reuse.
// Requires M % 32 == 0, K % 32 == 0. N may be ragged (clamped loads).
// ACT: 0=none, 1=silu. OUTMODE: 0=f32, 1=bf16, 2=both.
template<int ACT, bool HAS_BIAS, bool HAS_RES, int OUTMODE>
__global__ __launch_bounds__(256) void gemm_wmma_bf16(
    const __bf16* __restrict__ X, int ldx,
    const __bf16* __restrict__ W, int ldw,
    const float* __restrict__ bias,
    const float* __restrict__ res,
    float* __restrict__ Y, __bf16* __restrict__ Yb, int ldy,
    int M, int N, int K, float alpha)
{
    int wave = (blockIdx.x * blockDim.x + threadIdx.x) >> 5;
    int lane = threadIdx.x & 31;
    int mgroups = M >> 5;
    int ngroups = (N + 63) >> 6;
    if (wave >= mgroups * ngroups) return;
    int mg = wave / ngroups;
    int ng = wave - mg * ngroups;
    int lo = lane & 15, hi = lane >> 4;

    // A layout: k = hi*16 + i -> run [hi*16, +16)
    const __bf16* xr0 = X + (size_t)(mg * 32 + lo) * ldx + hi * 16;
    const __bf16* xr1 = xr0 + (size_t)16 * ldx;
    // B layout: k = hi*16 + i -> run [hi*16, +16)
    int nbase = ng * 64;
    const __bf16* wr0, *wr1, *wr2, *wr3;
    bool nok[4];
    {
        int n0 = nbase + lo,      n1 = nbase + 16 + lo;
        int n2 = nbase + 32 + lo, n3 = nbase + 48 + lo;
        nok[0] = n0 < N; nok[1] = n1 < N; nok[2] = n2 < N; nok[3] = n3 < N;
        wr0 = W + (size_t)(nok[0] ? n0 : (N - 1)) * ldw + hi * 16;
        wr1 = W + (size_t)(nok[1] ? n1 : (N - 1)) * ldw + hi * 16;
        wr2 = W + (size_t)(nok[2] ? n2 : (N - 1)) * ldw + hi * 16;
        wr3 = W + (size_t)(nok[3] ? n3 : (N - 1)) * ldw + hi * 16;
    }

    v8f acc[8] = {{}, {}, {}, {}, {}, {}, {}, {}};
    for (int kb = 0; kb < K; kb += 32) {
        v8bf a0l = *(const v8bf*)xr0, a0h = *(const v8bf*)(xr0 + 8);
        v8bf a1l = *(const v8bf*)xr1, a1h = *(const v8bf*)(xr1 + 8);
        v8bf b0l = *(const v8bf*)wr0, b0h = *(const v8bf*)(wr0 + 8);
        v8bf b1l = *(const v8bf*)wr1, b1h = *(const v8bf*)(wr1 + 8);
        v8bf b2l = *(const v8bf*)wr2, b2h = *(const v8bf*)(wr2 + 8);
        v8bf b3l = *(const v8bf*)wr3, b3h = *(const v8bf*)(wr3 + 8);
        v16bf af0 = cat16(a0l, a0h), af1 = cat16(a1l, a1h);
        v16bf bf0 = cat16(b0l, b0h), bf1 = cat16(b1l, b1h);
        v16bf bf2 = cat16(b2l, b2h), bf3 = cat16(b3l, b3h);
        acc[0] = __builtin_amdgcn_wmma_f32_16x16x32_bf16(false, af0, false, bf0, (short)0, acc[0], false, false);
        acc[1] = __builtin_amdgcn_wmma_f32_16x16x32_bf16(false, af1, false, bf0, (short)0, acc[1], false, false);
        acc[2] = __builtin_amdgcn_wmma_f32_16x16x32_bf16(false, af0, false, bf1, (short)0, acc[2], false, false);
        acc[3] = __builtin_amdgcn_wmma_f32_16x16x32_bf16(false, af1, false, bf1, (short)0, acc[3], false, false);
        acc[4] = __builtin_amdgcn_wmma_f32_16x16x32_bf16(false, af0, false, bf2, (short)0, acc[4], false, false);
        acc[5] = __builtin_amdgcn_wmma_f32_16x16x32_bf16(false, af1, false, bf2, (short)0, acc[5], false, false);
        acc[6] = __builtin_amdgcn_wmma_f32_16x16x32_bf16(false, af0, false, bf3, (short)0, acc[6], false, false);
        acc[7] = __builtin_amdgcn_wmma_f32_16x16x32_bf16(false, af1, false, bf3, (short)0, acc[7], false, false);
        xr0 += 32; xr1 += 32; wr0 += 32; wr1 += 32; wr2 += 32; wr3 += 32;
    }

#pragma unroll
    for (int j = 0; j < 4; ++j) {
        if (!nok[j]) continue;
        int n = nbase + j * 16 + lo;
        float bv = HAS_BIAS ? bias[n] : 0.0f;
#pragma unroll
        for (int mi = 0; mi < 2; ++mi) {
#pragma unroll
            for (int r = 0; r < 8; ++r) {
                int mo = mg * 32 + mi * 16 + hi * 8 + r;  // D: row = hi*8 + r
                float v = acc[j * 2 + mi][r] + bv;
                if (ACT == 1) v = silu_f(v);
                v *= alpha;
                if (HAS_RES) v += res[(size_t)mo * ldy + n];
                if (OUTMODE == 0 || OUTMODE == 2) Y[(size_t)mo * ldy + n] = v;
                if (OUTMODE == 1 || OUTMODE == 2) Yb[(size_t)mo * ldy + n] = (__bf16)v;
            }
        }
    }
}

// ---- dt_proj (K=8, too ragged for WMMA): dt = softplus(xdbl[:, :8] @ W^T + b)
__global__ __launch_bounds__(256) void dtproj_kernel(
    const float* __restrict__ xdbl, const float* __restrict__ w,
    const float* __restrict__ b, float* __restrict__ dt, int ntok)
{
    int idx = blockIdx.x * blockDim.x + threadIdx.x;
    if (idx >= ntok * DINNER) return;
    int d = idx & (DINNER - 1);
    int tok = idx >> 8;
    const float* xr = xdbl + (size_t)tok * 264;
    const float* wr = w + d * 8;
    float acc = b[d];
#pragma unroll
    for (int j = 0; j < 8; ++j) acc += xr[j] * wr[j];
    dt[(size_t)tok * DINNER + d] = (acc > 20.0f) ? acc : __logf(1.0f + __expf(acc));
}

// ---- Causal depthwise conv (width 4) + SiLU; writes f32 (scan) + bf16 (GEMM)
__global__ __launch_bounds__(256) void conv_silu_kernel(
    const float* __restrict__ xz, const float* __restrict__ cw,
    const float* __restrict__ cb, float* __restrict__ xi,
    __bf16* __restrict__ xib, int ntok)
{
    int idx = blockIdx.x * blockDim.x + threadIdx.x;
    if (idx >= ntok * DINNER) return;
    int d = idx & (DINNER - 1);
    int tok = idx >> 8;
    int t = tok & (LSEQ - 1);
    float acc = cb[d];
#pragma unroll
    for (int j = 0; j < 4; ++j) {
        int tt = t - 3 + j;
        float xv = (tt >= 0) ? xz[(size_t)(tok - 3 + j) * 512 + d] : 0.0f;
        acc += cw[d * 4 + j] * xv;
    }
    float v = silu_f(acc);
    xi[(size_t)tok * DINNER + d] = v;
    xib[(size_t)tok * DINNER + d] = (__bf16)v;
}

// ---- Selective scan: one wave per (b,d); 4 states/lane over n=0..127 ----
__global__ __launch_bounds__(256) void scan_kernel(
    const float* __restrict__ dt, const float* __restrict__ xdbl,
    const float* __restrict__ xi, const float* __restrict__ xz,
    const float* __restrict__ A_log, const float* __restrict__ Dvec,
    __bf16* __restrict__ y)
{
    int wave = (blockIdx.x * blockDim.x + threadIdx.x) >> 5;
    int lane = threadIdx.x & 31;
    if (wave >= BATCH * DINNER) return;
    int b = wave >> 8;
    int d = wave & (DINNER - 1);

    const float* arow = A_log + (size_t)d * DSTATE;
    float a0 = -__expf(arow[lane]);
    float a1 = -__expf(arow[lane + 32]);
    float a2 = -__expf(arow[lane + 64]);
    float a3 = -__expf(arow[lane + 96]);
    float Dd = Dvec[d];
    float h0 = 0.f, h1 = 0.f, h2 = 0.f, h3 = 0.f;

    size_t tok = (size_t)b * LSEQ;
    for (int t = 0; t < LSEQ; ++t, ++tok) {
        float dtv = dt[tok * DINNER + d];
        float u   = xi[tok * DINNER + d];
        const float* xr = xdbl + tok * 264;
        float du = dtv * u;
        float B0 = xr[8 + lane],   C0 = xr[136 + lane];
        float B1 = xr[40 + lane],  C1 = xr[168 + lane];
        float B2 = xr[72 + lane],  C2 = xr[200 + lane];
        float B3 = xr[104 + lane], C3 = xr[232 + lane];
        h0 = __expf(dtv * a0) * h0 + du * B0;
        h1 = __expf(dtv * a1) * h1 + du * B1;
        h2 = __expf(dtv * a2) * h2 + du * B2;
        h3 = __expf(dtv * a3) * h3 + du * B3;
        float p = h0 * C0 + h1 * C1 + h2 * C2 + h3 * C3;
#pragma unroll
        for (int off = 16; off; off >>= 1) p += __shfl_xor(p, off, 32);
        if (lane == 0) {
            float z = xz[tok * 512 + DINNER + d];
            y[tok * DINNER + d] = (__bf16)((p + u * Dd) * silu_f(z));
        }
    }
}

static inline int gemm_blocks(int M, int N)
{
    int tiles = (M >> 5) * ((N + 63) >> 6);
    return (tiles + 7) / 8;   // 8 waves / 256-thread block
}

extern "C" void kernel_launch(void* const* d_in, const int* in_sizes, int n_in,
                              void* d_out, int out_size, void* d_ws, size_t ws_size,
                              hipStream_t stream)
{
    const float* x        = (const float*)d_in[0];
    const float* ln1_g    = (const float*)d_in[1];
    const float* ln1_b    = (const float*)d_in[2];
    const float* ff1_w1   = (const float*)d_in[3];
    const float* ff1_b1   = (const float*)d_in[4];
    const float* ff1_w2   = (const float*)d_in[5];
    const float* ff1_b2   = (const float*)d_in[6];
    const float* in_w     = (const float*)d_in[7];
    const float* in_b     = (const float*)d_in[8];
    const float* conv_w   = (const float*)d_in[9];
    const float* conv_b   = (const float*)d_in[10];
    const float* xproj_w  = (const float*)d_in[11];
    const float* dtproj_w = (const float*)d_in[12];
    const float* dtproj_b = (const float*)d_in[13];
    const float* A_log    = (const float*)d_in[14];
    const float* Dvec     = (const float*)d_in[15];
    const float* out_w    = (const float*)d_in[16];
    const float* out_b    = (const float*)d_in[17];
    const float* ln2_g    = (const float*)d_in[18];
    const float* ln2_b    = (const float*)d_in[19];
    const float* ff2_w1   = (const float*)d_in[20];
    const float* ff2_b1   = (const float*)d_in[21];
    const float* ff2_w2   = (const float*)d_in[22];
    const float* ff2_b2   = (const float*)d_in[23];
    const float* pn_g     = (const float*)d_in[24];
    const float* pn_b     = (const float*)d_in[25];
    float* out = (float*)d_out;

    // ---- workspace layout: f32 region then bf16 region ----
    float* ws    = (float*)d_ws;
    float* wsXz  = ws;                                  // NTOK*512  xz
    float* wsX1  = wsXz + (size_t)NTOK * 512;           // NTOK*128  x after block1 / block3 out
    float* wsXi  = wsX1 + (size_t)NTOK * 128;           // NTOK*256  conv+silu (f32 for scan)
    float* wsXd  = wsXi + (size_t)NTOK * 256;           // NTOK*264  x_dbl
    float* wsDt  = wsXd + (size_t)NTOK * 264;           // NTOK*256  dt
    float* wsX2  = wsDt + (size_t)NTOK * 256;           // NTOK*128  x after mamba
    __bf16* hb    = (__bf16*)(wsX2 + (size_t)NTOK * 128);
    __bf16* hbLn  = hb;                                 // NTOK*128
    __bf16* hbHid = hbLn  + (size_t)NTOK * 128;         // NTOK*512
    __bf16* hbX1  = hbHid + (size_t)NTOK * 512;         // NTOK*128
    __bf16* hbXi  = hbX1  + (size_t)NTOK * 128;         // NTOK*256
    __bf16* hbY   = hbXi  + (size_t)NTOK * 256;         // NTOK*256
    __bf16* wb11  = hbY   + (size_t)NTOK * 256;         // ff1_w1 512x128
    __bf16* wb12  = wb11 + 512 * 128;                   // ff1_w2 128x512
    __bf16* wbIn  = wb12 + 128 * 512;                   // in_w   512x128
    __bf16* wbXp  = wbIn + 512 * 128;                   // xproj  264x256
    __bf16* wbOut = wbXp + 264 * 256;                   // out_w  128x256
    __bf16* wb21  = wbOut + 128 * 256;                  // ff2_w1 512x128
    __bf16* wb22  = wb21 + 512 * 128;                   // ff2_w2 128x512

    // ---- one-shot weight conversion (tiny) ----
    auto cvt = [&](const float* s, __bf16* d, int n) {
        cvt_bf16_kernel<<<(n + 255) / 256, 256, 0, stream>>>(s, d, n);
    };
    cvt(ff1_w1, wb11, 512 * 128);
    cvt(ff1_w2, wb12, 128 * 512);
    cvt(in_w,   wbIn, 512 * 128);
    cvt(xproj_w, wbXp, 264 * 256);
    cvt(out_w,  wbOut, 128 * 256);
    cvt(ff2_w1, wb21, 512 * 128);
    cvt(ff2_w2, wb22, 128 * 512);

    // ---- Block 1: x1 = 0.5 * FF(LN(x)) + x ----
    ln_kernel<true><<<NTOK / 8, 256, 0, stream>>>(x, ln1_g, ln1_b, nullptr, hbLn, NTOK);
    gemm_wmma_bf16<1, true, false, 1><<<gemm_blocks(NTOK, 512), 256, 0, stream>>>(
        hbLn, 128, wb11, 128, ff1_b1, nullptr, nullptr, hbHid, 512, NTOK, 512, 128, 1.0f);
    gemm_wmma_bf16<0, true, true, 2><<<gemm_blocks(NTOK, 128), 256, 0, stream>>>(
        hbHid, 512, wb12, 512, ff1_b2, x, wsX1, hbX1, 128, NTOK, 128, 512, 0.5f);

    // ---- Mamba ----
    gemm_wmma_bf16<0, true, false, 0><<<gemm_blocks(NTOK, 512), 256, 0, stream>>>(
        hbX1, 128, wbIn, 128, in_b, nullptr, wsXz, nullptr, 512, NTOK, 512, 128, 1.0f);
    conv_silu_kernel<<<(NTOK * DINNER) / 256, 256, 0, stream>>>(wsXz, conv_w, conv_b, wsXi, hbXi, NTOK);
    gemm_wmma_bf16<0, false, false, 0><<<gemm_blocks(NTOK, 264), 256, 0, stream>>>(
        hbXi, 256, wbXp, 256, nullptr, nullptr, wsXd, nullptr, 264, NTOK, 264, 256, 1.0f);
    dtproj_kernel<<<(NTOK * DINNER) / 256, 256, 0, stream>>>(wsXd, dtproj_w, dtproj_b, wsDt, NTOK);
    scan_kernel<<<(BATCH * DINNER) / 8, 256, 0, stream>>>(wsDt, wsXd, wsXi, wsXz, A_log, Dvec, hbY);
    gemm_wmma_bf16<0, true, true, 0><<<gemm_blocks(NTOK, 128), 256, 0, stream>>>(
        hbY, 256, wbOut, 256, out_b, wsX1, wsX2, nullptr, 128, NTOK, 128, 256, 1.0f);

    // ---- Block 3: x3 = 0.5 * FF(LN(x2)) + x2 ----
    ln_kernel<true><<<NTOK / 8, 256, 0, stream>>>(wsX2, ln2_g, ln2_b, nullptr, hbLn, NTOK);
    gemm_wmma_bf16<1, true, false, 1><<<gemm_blocks(NTOK, 512), 256, 0, stream>>>(
        hbLn, 128, wb21, 128, ff2_b1, nullptr, nullptr, hbHid, 512, NTOK, 512, 128, 1.0f);
    gemm_wmma_bf16<0, true, true, 0><<<gemm_blocks(NTOK, 128), 256, 0, stream>>>(
        hbHid, 512, wb22, 512, ff2_b2, wsX2, wsX1, nullptr, 128, NTOK, 128, 512, 0.5f);

    // ---- final LN -> out ----
    ln_kernel<false><<<NTOK / 8, 256, 0, stream>>>(wsX1, pn_g, pn_b, out, nullptr, NTOK);
}
